// LSTM_13176959664285
// MI455X (gfx1250) — matrix operand order repaired and checked by reading, not compiled, not run on hardware
//
#include <hip/hip_runtime.h>

// ---------------------------------------------------------------------------
// 2-layer LSTM (B=128, T=2048, H=256) + masked mean pool + linear(7)
// MI455X / gfx1250 design:
//   * latency-bound recurrence -> 8 persistent workgroups (one per 16-row
//     batch chunk), both layers advanced per timestep (no h1 sequence in HBM)
//   * bf16 WMMA 16x16x32, f32 accumulate; weights pre-packed into per-lane
//     B-fragment order and streamed from L2 (192MB L2 >> 1.5MB weights)
//   * h-state in LDS (padded rows), c-state/gates/pool entirely in registers:
//     each wave owns the i,f,g,o gate tiles of one 16-wide hidden block
//   * K loops use unroll-2 (not full unroll) so the B-fragment stream stays
//     in VGPRs -- full unroll made LLVM hoist every load and spill to scratch
//   * gate nonlinearities use v_exp_f32 + v_rcp_f32 (no IEEE divide chains on
//     the serial critical path between the two GEMMs of each timestep)
// ---------------------------------------------------------------------------

typedef unsigned short u16;
typedef __attribute__((ext_vector_type(4)))  unsigned int v4u;
typedef __attribute__((ext_vector_type(16))) __bf16       v16bf;
typedef __attribute__((ext_vector_type(8)))  float        v8f;

union FragB16 { v4u u[2]; v16bf v; };

#define NB   128
#define NT   2048
#define NH   256
#define NG   1024          // 4*H gate columns
#define HSTR 264           // padded LDS row stride (halves) -> bank spread

__device__ __forceinline__ u16 f2bf(float f) {
  unsigned u = __float_as_uint(f);
  u += 0x7FFFu + ((u >> 16) & 1u);          // round-to-nearest-even
  return (u16)(u >> 16);
}
// sigmoid via v_exp_f32 + v_rcp_f32 (fast path, no div expansion)
__device__ __forceinline__ float sigf(float x) {
  return __builtin_amdgcn_rcpf(1.0f + __expf(-x));
}
// tanh(x) = 2*sigmoid(2x) - 1  (one exp + one rcp + one fma)
__device__ __forceinline__ float tanhf_(float x) {
  return 2.0f * __builtin_amdgcn_rcpf(1.0f + __expf(-2.0f * x)) - 1.0f;
}

#define WMMA(ACC, AV, BV)                                                       \
  ACC = __builtin_amdgcn_wmma_f32_16x16x32_bf16(false, AV, false, BV,           \
                                                (short)0, ACC, false, false)

// ---------------------------------------------------------------------------
// Kernel 0: pack weights into WMMA B-fragment order (bf16) + fuse biases.
//   W0p: [ntile 64][ktile 8][lane 32][16 halves]  from W_hh0  (K=256)
//   W1p: [ntile 64][ktile 16][lane 32][16 halves] from [W_ih1;W_hh1] (K=512)
//   lane<16: column n = nt*16+lane, K kt*32+0..15 ; lane>=16: K kt*32+16..31
// ---------------------------------------------------------------------------
__global__ __launch_bounds__(256) void prep_pack(
    const float* __restrict__ Whh0, const float* __restrict__ Wih1,
    const float* __restrict__ Whh1,
    const float* __restrict__ bih0, const float* __restrict__ bhh0,
    const float* __restrict__ bih1, const float* __restrict__ bhh1,
    u16* __restrict__ W0p, u16* __restrict__ W1p,
    float* __restrict__ b0, float* __restrict__ b1)
{
  int tid = blockIdx.x * blockDim.x + threadIdx.x;
  if (tid < 16384) {                              // W0 fragments
    int lane = tid & 31, kt = (tid >> 5) & 7, nt = tid >> 8;
    int n  = nt * 16 + (lane & 15);
    int kb = kt * 32 + (lane >> 4) * 16;
    const float* src = Whh0 + n * NH + kb;
    u16* dst = W0p + tid * 16;
#pragma unroll
    for (int h = 0; h < 16; ++h) dst[h] = f2bf(src[h]);
  } else if (tid < 16384 + 32768) {               // W1 fragments (K=512 concat)
    int id = tid - 16384;
    int lane = id & 31, kt = (id >> 5) & 15, nt = id >> 9;
    int n  = nt * 16 + (lane & 15);
    int kb = kt * 32 + (lane >> 4) * 16;
    const float* src = (kb < NH) ? (Wih1 + n * NH + kb)
                                 : (Whh1 + n * NH + (kb - NH));
    u16* dst = W1p + id * 16;
#pragma unroll
    for (int h = 0; h < 16; ++h) dst[h] = f2bf(src[h]);
  } else if (tid < 16384 + 32768 + 1024) {        // fused bias layer 0
    int i = tid - (16384 + 32768);
    b0[i] = bih0[i] + bhh0[i];
  } else if (tid < 16384 + 32768 + 2048) {        // fused bias layer 1
    int i = tid - (16384 + 32768 + 1024);
    b1[i] = bih1[i] + bhh1[i];
  }
}

// ---------------------------------------------------------------------------
// Kernel 1: persistent fused 2-layer LSTM recurrence + masked pooling.
// Grid = 8 blocks (one per 16 batch rows), 512 threads = 16 waves.
// Wave w owns hidden block j in [16w,16w+16): gate N-tiles {w,w+16,w+32,w+48}.
// ---------------------------------------------------------------------------
__global__ __launch_bounds__(512) void lstm_persist(
    const float* __restrict__ x, const int* __restrict__ lengths,
    const float* __restrict__ wih0,
    const u16* __restrict__ W0p, const u16* __restrict__ W1p,
    const float* __restrict__ b0, const float* __restrict__ b1,
    float* __restrict__ pooled)
{
  __shared__ alignas(16) u16 hbuf1[16 * HSTR];   // h1 state, bf16
  __shared__ alignas(16) u16 hbuf2[16 * HSTR];   // h2 state, bf16
  __shared__ float xstage[16];                   // x[:,t] for this chunk

  const int tid  = threadIdx.x;
  const int w    = tid >> 5;       // wave id 0..15
  const int lane = tid & 31;
  const int jl   = lane & 15;      // N within tile
  const int hi   = lane >> 4;      // lane half
  const int brow = blockIdx.x * 16;

  for (int i = tid; i < 16 * HSTR; i += 512) { hbuf1[i] = 0; hbuf2[i] = 0; }
  if (tid < 16) xstage[tid] = x[(brow + tid) * NT];

  // warm L2 with packed weights (global_prefetch_b8), one line per iteration
  for (int off = tid; off < 8192; off += 512)
    __builtin_prefetch(W0p + off * 32, 0, 1);
  for (int off = tid; off < 16384; off += 512)
    __builtin_prefetch(W1p + off * 32, 0, 1);

  // wave-uniform fragment base pointers (SGPR base + one shared lane offset)
  const v4u* base0[4];   // layer 0: + kt*64 + 2*lane
  const v4u* base1[4];   // layer 1: + kt*64 + 2*lane
#pragma unroll
  for (int g = 0; g < 4; ++g) {
    base0[g] = (const v4u*)W0p + (size_t)(g * 16 + w) * 512;   // 8 kt * 64
    base1[g] = (const v4u*)W1p + (size_t)(g * 16 + w) * 1024;  // 16 kt * 64
  }
  const int lo = 2 * lane;

  // per-wave constants: input weight + biases for the 4 owned gate columns
  float wv0[4], bv0[4], bv1[4];
#pragma unroll
  for (int g = 0; g < 4; ++g) {
    int n = g * NH + w * 16 + jl;
    wv0[g] = wih0[n];
    bv0[g] = b0[n];
    bv1[g] = b1[n];
  }
  int lenv[8];
#pragma unroll
  for (int r = 0; r < 8; ++r) lenv[r] = lengths[brow + r + 8 * hi];

  v8f c1   = {0,0,0,0,0,0,0,0};
  v8f c2   = {0,0,0,0,0,0,0,0};
  v8f pool = {0,0,0,0,0,0,0,0};
  __syncthreads();

  const int abase = jl * HSTR + hi * 16;  // A-frag base (contiguous k = 16*hi + s)

  for (int t = 0; t < NT; ++t) {
    // ---------------- layer 0: gates = x*Wih0 + b0 + h1 @ Whh0^T ----------
    float xm[8];
#pragma unroll
    for (int r = 0; r < 8; ++r) xm[r] = xstage[r + 8 * hi];
    v8f a0, a1, a2, a3;
#pragma unroll
    for (int r = 0; r < 8; ++r) {
      a0[r] = xm[r] * wv0[0] + bv0[0];
      a1[r] = xm[r] * wv0[1] + bv0[1];
      a2[r] = xm[r] * wv0[2] + bv0[2];
      a3[r] = xm[r] * wv0[3] + bv0[3];
    }
#pragma unroll 2
    for (int kt = 0; kt < 8; ++kt) {
      FragB16 A;
      A.u[0] = *(const v4u*)(&hbuf1[abase + kt * 32]);
      A.u[1] = *(const v4u*)(&hbuf1[abase + kt * 32 + 8]);
      FragB16 B0, B1, B2, B3;
      B0.u[0] = base0[0][kt * 64 + lo]; B0.u[1] = base0[0][kt * 64 + lo + 1];
      B1.u[0] = base0[1][kt * 64 + lo]; B1.u[1] = base0[1][kt * 64 + lo + 1];
      B2.u[0] = base0[2][kt * 64 + lo]; B2.u[1] = base0[2][kt * 64 + lo + 1];
      B3.u[0] = base0[3][kt * 64 + lo]; B3.u[1] = base0[3][kt * 64 + lo + 1];
      WMMA(a0, A.v, B0.v);
      WMMA(a1, A.v, B1.v);
      WMMA(a2, A.v, B2.v);
      WMMA(a3, A.v, B3.v);
    }
    __syncthreads();   // all A-frag reads of hbuf1 done

    // cell update layer 0 (all in registers; D elem = (M=r+8*hi, N=jl))
    {
      v8f hnew;
#pragma unroll
      for (int r = 0; r < 8; ++r) {
        float ig = sigf(a0[r]);
        float fg = sigf(a1[r]);
        float gg = tanhf_(a2[r]);
        float og = sigf(a3[r]);
        c1[r] = fg * c1[r] + ig * gg;
        hnew[r] = og * tanhf_(c1[r]);
        hbuf1[(r + 8 * hi) * HSTR + w * 16 + jl] = f2bf(hnew[r]);
      }
    }
    __syncthreads();   // hbuf1 now holds h1[t]

    // ------------- layer 1: gates = b1 + [h1[t];h2[t-1]] @ W1^T (K=512) ----
#pragma unroll
    for (int r = 0; r < 8; ++r) {
      a0[r] = bv1[0]; a1[r] = bv1[1]; a2[r] = bv1[2]; a3[r] = bv1[3];
    }
#pragma unroll 2
    for (int kt = 0; kt < 8; ++kt) {           // K 0..255 from h1[t]
      FragB16 A;
      A.u[0] = *(const v4u*)(&hbuf1[abase + kt * 32]);
      A.u[1] = *(const v4u*)(&hbuf1[abase + kt * 32 + 8]);
      FragB16 B0, B1, B2, B3;
      B0.u[0] = base1[0][kt * 64 + lo]; B0.u[1] = base1[0][kt * 64 + lo + 1];
      B1.u[0] = base1[1][kt * 64 + lo]; B1.u[1] = base1[1][kt * 64 + lo + 1];
      B2.u[0] = base1[2][kt * 64 + lo]; B2.u[1] = base1[2][kt * 64 + lo + 1];
      B3.u[0] = base1[3][kt * 64 + lo]; B3.u[1] = base1[3][kt * 64 + lo + 1];
      WMMA(a0, A.v, B0.v);
      WMMA(a1, A.v, B1.v);
      WMMA(a2, A.v, B2.v);
      WMMA(a3, A.v, B3.v);
    }
#pragma unroll 2
    for (int kt = 8; kt < 16; ++kt) {          // K 256..511 from h2[t-1]
      FragB16 A;
      A.u[0] = *(const v4u*)(&hbuf2[abase + (kt - 8) * 32]);
      A.u[1] = *(const v4u*)(&hbuf2[abase + (kt - 8) * 32 + 8]);
      FragB16 B0, B1, B2, B3;
      B0.u[0] = base1[0][kt * 64 + lo]; B0.u[1] = base1[0][kt * 64 + lo + 1];
      B1.u[0] = base1[1][kt * 64 + lo]; B1.u[1] = base1[1][kt * 64 + lo + 1];
      B2.u[0] = base1[2][kt * 64 + lo]; B2.u[1] = base1[2][kt * 64 + lo + 1];
      B3.u[0] = base1[3][kt * 64 + lo]; B3.u[1] = base1[3][kt * 64 + lo + 1];
      WMMA(a0, A.v, B0.v);
      WMMA(a1, A.v, B1.v);
      WMMA(a2, A.v, B2.v);
      WMMA(a3, A.v, B3.v);
    }
    __syncthreads();   // all reads of hbuf1/hbuf2 done

    // cell update layer 1 + masked pooling accumulate
    {
      v8f hnew;
#pragma unroll
      for (int r = 0; r < 8; ++r) {
        float ig = sigf(a0[r]);
        float fg = sigf(a1[r]);
        float gg = tanhf_(a2[r]);
        float og = sigf(a3[r]);
        c2[r] = fg * c2[r] + ig * gg;
        hnew[r] = og * tanhf_(c2[r]);
        hbuf2[(r + 8 * hi) * HSTR + w * 16 + jl] = f2bf(hnew[r]);
        if (t < lenv[r]) pool[r] += hnew[r];
      }
    }
    if (tid < 16 && (t + 1) < NT)
      xstage[tid] = x[(brow + tid) * NT + t + 1];
    __syncthreads();   // h2[t], x[t+1] visible for next iteration
  }

  // write pooled mean (f32) for this chunk's hidden block
#pragma unroll
  for (int r = 0; r < 8; ++r) {
    int m = r + 8 * hi;
    pooled[(brow + m) * NH + w * 16 + jl] = pool[r] / (float)lenv[r];
  }
}

// ---------------------------------------------------------------------------
// Kernel 2: tiny final linear: out[b][c] = pooled[b] . W_lin[c] + b_lin[c]
// ---------------------------------------------------------------------------
__global__ __launch_bounds__(128) void final_linear(
    const float* __restrict__ pooled, const float* __restrict__ Wlin,
    const float* __restrict__ blin, float* __restrict__ out)
{
  int tid = blockIdx.x * blockDim.x + threadIdx.x;
  if (tid >= NB * 7) return;
  int b = tid / 7, cc = tid % 7;
  const float* p  = pooled + b * NH;
  const float* wr = Wlin + cc * NH;
  float s = blin[cc];
#pragma unroll 8
  for (int j = 0; j < NH; ++j) s += p[j] * wr[j];
  out[b * 7 + cc] = s;
}

// ---------------------------------------------------------------------------
extern "C" void kernel_launch(void* const* d_in, const int* in_sizes, int n_in,
                              void* d_out, int out_size, void* d_ws, size_t ws_size,
                              hipStream_t stream)
{
  (void)in_sizes; (void)n_in; (void)out_size;

  const float* x     = (const float*)d_in[0];
  const int*   lens  = (const int*)d_in[1];
  const float* Wih0  = (const float*)d_in[2];   // [1024,1]
  const float* Whh0  = (const float*)d_in[3];   // [1024,256]
  const float* bih0  = (const float*)d_in[4];
  const float* bhh0  = (const float*)d_in[5];
  const float* Wih1  = (const float*)d_in[6];   // [1024,256]
  const float* Whh1  = (const float*)d_in[7];   // [1024,256]
  const float* bih1  = (const float*)d_in[8];
  const float* bhh1  = (const float*)d_in[9];
  const float* Wlin  = (const float*)d_in[10];  // [7,256]
  const float* blin  = (const float*)d_in[11];  // [7]

  // workspace layout (~1.72 MB total)
  char* ws = (char*)d_ws;
  size_t o = 0;
  u16*   W0p    = (u16*)(ws + o);  o += (size_t)16384 * 32;   // 512 KB
  u16*   W1p    = (u16*)(ws + o);  o += (size_t)32768 * 32;   // 1 MB
  float* b0     = (float*)(ws + o); o += 1024 * sizeof(float);
  float* b1     = (float*)(ws + o); o += 1024 * sizeof(float);
  float* pooled = (float*)(ws + o); o += (size_t)NB * NH * sizeof(float);
  (void)ws_size;  // requires ws_size >= ~1.72 MB

  prep_pack<<<200, 256, 0, stream>>>(Whh0, Wih1, Whh1, bih0, bhh0, bih1, bhh1,
                                     W0p, W1p, b0, b1);
  lstm_persist<<<NB / 16, 512, 0, stream>>>(x, lens, Wih0, W0p, W1p, b0, b1,
                                            pooled);
  final_linear<<<7, 128, 0, stream>>>(pooled, Wlin, blin, (float*)d_out);
}
